// Enhanced_DCN_SCC_Conv_88270167867683
// MI455X (gfx1250) — compile-verified, matrix operand order checked by reading, not run on hardware
//
#include <hip/hip_runtime.h>
#include <math.h>

typedef __attribute__((ext_vector_type(16))) _Float16 v16h;
typedef __attribute__((ext_vector_type(8)))  _Float16 v8h;
typedef __attribute__((ext_vector_type(8)))  float    v8f;

// Problem constants (from setup_inputs)
#define BN_  4
#define C1_  128
#define CH_  128
#define C2_  256
#define H_   64
#define W_   64
#define HW_  (H_ * W_)
#define EPS_ 1e-5f

__device__ __forceinline__ int clampi(int v, int lo, int hi) {
    return v < lo ? lo : (v > hi ? hi : v);
}
__device__ __forceinline__ float sigmoidf_(float v) {
    return 1.0f / (1.0f + __expf(-v));
}

// LDS byte-offset of a __shared__ object (generic -> AS(3) cast strips aperture).
__device__ __forceinline__ unsigned int lds_addr_of(const void* p) {
    return (unsigned int)(unsigned long long)(__attribute__((address_space(3))) const void*)p;
}
// CDNA5 async global->LDS copy, 16 bytes per lane (tracked by ASYNCcnt).
__device__ __forceinline__ void async_ld_b128(unsigned int lds_off, const void* gptr) {
    asm volatile("global_load_async_to_lds_b128 %0, %1, off"
                 :: "v"(lds_off), "v"(gptr) : "memory");
}
__device__ __forceinline__ void wait_async0() {
    asm volatile("s_wait_asynccnt 0" ::: "memory");
}

// ---------------------------------------------------------------------------
// Offset convolution: plain direct conv, few output channels (18 or 50).
// ---------------------------------------------------------------------------
template <int K>
__global__ void offset_conv_kernel(const float* __restrict__ x,
                                   const float* __restrict__ w,
                                   float* __restrict__ out, int OC) {
    const int P = K / 2;
    int idx = blockIdx.x * blockDim.x + threadIdx.x;
    int total = BN_ * OC * HW_;
    if (idx >= total) return;
    int wp = idx % W_;
    int t  = idx / W_;
    int hp = t % H_; t /= H_;
    int oc = t % OC;
    int b  = t / OC;

    float acc = 0.0f;
    for (int c = 0; c < C1_; ++c) {
        const float* xp = x + ((size_t)(b * C1_ + c)) * HW_;
        const float* wq = w + ((size_t)(oc * C1_ + c)) * K * K;
        for (int ky = 0; ky < K; ++ky) {
            int yy = hp + ky - P;
            if (yy < 0 || yy >= H_) continue;
            for (int kx = 0; kx < K; ++kx) {
                int xx = wp + kx - P;
                if (xx < 0 || xx >= W_) continue;
                acc += xp[yy * W_ + xx] * wq[ky * K + kx];
            }
        }
    }
    out[idx] = acc;
}

// ---------------------------------------------------------------------------
// One-time weight repack: fp32 (CH,C1,K,K) -> f16 in WMMA A-fragment lane
// order, indexed [tap][cc][otile][lane][j] so the hot loop does one aligned
// v16h (32B) load per lane.
// ---------------------------------------------------------------------------
template <int K>
__global__ void pack_dcn_w_kernel(const float* __restrict__ wgt,
                                  _Float16* __restrict__ out) {
    constexpr int K2 = K * K;
    int i = blockIdx.x * 256 + threadIdx.x;
    int total = K2 * 4 * 8 * 32 * 16;
    if (i >= total) return;
    int j   = i & 15;
    int ln  = (i >> 4) & 31;
    int ot  = (i >> 9) & 7;
    int cc  = (i >> 12) & 3;
    int tap = i >> 14;
    int mn = ln & 15;
    int kb = (j < 8) ? ((ln < 16) ? 0 : 8) : ((ln < 16) ? 16 : 24);
    int kk = kb + (j & 7);
    out[i] = (_Float16)wgt[((size_t)(ot * 16 + mn) * C1_ + cc * 32 + kk) * K2 + tap];
}

__global__ void pack_fuse_w_kernel(const float* __restrict__ wgt,  // (256,256)
                                   _Float16* __restrict__ out) {
    int i = blockIdx.x * 256 + threadIdx.x;
    if (i >= 8 * 16 * 32 * 16) return;
    int j  = i & 15;
    int ln = (i >> 4) & 31;
    int ot = (i >> 9) & 15;
    int cc = i >> 13;
    int mn = ln & 15;
    int kb = (j < 8) ? ((ln < 16) ? 0 : 8) : ((ln < 16) ? 16 : 24);
    int kk = kb + (j & 7);
    out[i] = (_Float16)wgt[(size_t)(ot * 16 + mn) * C2_ + cc * 32 + kk];
}

// Pack attended branch outputs into pixel-major f16 comb: (B, HW, 256)
__global__ void pack_comb_kernel(const float* __restrict__ f1,
                                 const float* __restrict__ f2,
                                 _Float16* __restrict__ combh) {
    int i = blockIdx.x * 256 + threadIdx.x;   // B*HW*C2
    if (i >= BN_ * HW_ * C2_) return;
    int c = i & (C2_ - 1);
    int t = i >> 8;
    int p = t & (HW_ - 1);
    int b = t >> 12;
    float v = (c < CH_) ? f1[((size_t)(b * CH_ + c)) * HW_ + p]
                        : f2[((size_t)(b * CH_ + (c - CH_))) * HW_ + p];
    combh[i] = (_Float16)v;
}

// ---------------------------------------------------------------------------
// Deformable conv as implicit GEMM on WMMA (f16 in, f32 acc) + BN + SiLU.
// Block = 256 threads = 8 waves; wave w owns o-channels [16w,16w+16) for one
// 16-pixel strip. K-dim = C1*K2, chunked by 32.
// ---------------------------------------------------------------------------
template <int K>
__global__ __launch_bounds__(256) void dcn_wmma_kernel(
    const float* __restrict__ x,       // (B,C1,H,W)
    const float* __restrict__ off,     // (B,2*K2,H,W)
    const _Float16* __restrict__ wp,   // packed f16 weights
    const float* __restrict__ bn_g, const float* __restrict__ bn_b,
    const float* __restrict__ bn_m, const float* __restrict__ bn_v,
    float* __restrict__ out)           // (B,CH,H,W)
{
    constexpr int K2 = K * K;
    constexpr int P  = K / 2;

    int tile = blockIdx.x;               // B*H*(W/16) = 1024
    int wseg = tile & 3;
    int h    = (tile >> 2) & (H_ - 1);
    int b    = tile >> 8;
    int w0   = wseg * 16;

    int tid  = threadIdx.x;
    int lane = tid & 31;
    int wave = tid >> 5;

    __shared__ int   s_y0[16];
    __shared__ int   s_x0[16];
    __shared__ float s_bw[16][4];
    __shared__ __align__(64) _Float16 s_Bt[16][32];   // [pixel][channel]

    v8f acc = {0.f, 0.f, 0.f, 0.f, 0.f, 0.f, 0.f, 0.f};

    const int mn  = lane & 15;
    const int kb0 = (lane < 16) ? 0 : 8;
    const int kb1 = (lane < 16) ? 16 : 24;

    for (int tap = 0; tap < K2; ++tap) {
        int ky = tap / K, kx = tap % K;
        if (tid < 16) {
            int p = tid;
            int base = ((b * (2 * K2) + 2 * tap) * H_ + h) * W_ + (w0 + p);
            float dy = off[base];
            float dx = off[base + HW_];
            float yf = (float)(h - P + ky) + dy;
            float xf = (float)(w0 + p - P + kx) + dx;
            float y0f = floorf(yf), x0f = floorf(xf);
            float ly = yf - y0f, lx = xf - x0f;
            int y0 = (int)y0f, x0 = (int)x0f;
            float vy0 = (y0 >= 0 && y0 <= H_ - 1) ? 1.f : 0.f;
            float vy1 = (y0 + 1 >= 0 && y0 + 1 <= H_ - 1) ? 1.f : 0.f;
            float vx0 = (x0 >= 0 && x0 <= W_ - 1) ? 1.f : 0.f;
            float vx1 = (x0 + 1 >= 0 && x0 + 1 <= W_ - 1) ? 1.f : 0.f;
            s_y0[p] = y0; s_x0[p] = x0;
            s_bw[p][0] = (1.f - ly) * (1.f - lx) * vy0 * vx0;
            s_bw[p][1] = (1.f - ly) * lx         * vy0 * vx1;
            s_bw[p][2] = ly * (1.f - lx)         * vy1 * vx0;
            s_bw[p][3] = ly * lx                 * vy1 * vx1;
        }
        __syncthreads();

        for (int cc = 0; cc < C1_ / 32; ++cc) {
            // Cooperative bilinear sampling of B-chunk S^T[16 x 32].
            {
                int p    = tid & 15;
                int crel = (tid >> 4) * 2;
                int y0 = s_y0[p], x0 = s_x0[p];
                int yc0 = clampi(y0, 0, H_ - 1), yc1 = clampi(y0 + 1, 0, H_ - 1);
                int xc0 = clampi(x0, 0, W_ - 1), xc1 = clampi(x0 + 1, 0, W_ - 1);
                float w00 = s_bw[p][0], w01 = s_bw[p][1];
                float w10 = s_bw[p][2], w11 = s_bw[p][3];
#pragma unroll
                for (int q = 0; q < 2; ++q) {
                    int c = cc * 32 + crel + q;
                    const float* xp = x + ((size_t)(b * C1_ + c)) * HW_;
                    float v = xp[yc0 * W_ + xc0] * w00 + xp[yc0 * W_ + xc1] * w01 +
                              xp[yc1 * W_ + xc0] * w10 + xp[yc1 * W_ + xc1] * w11;
                    s_Bt[p][crel + q] = (_Float16)v;
                }
            }
            __syncthreads();

            // A frag: one aligned 32B v16h load of pre-packed weights.
            const _Float16* ap =
                wp + (((size_t)((tap * 4 + cc) * 8 + wave) * 32 + lane) << 4);
            __builtin_prefetch(ap + (size_t)8 * 32 * 16, 0, 3);
            v16h aF = *(const v16h*)ap;

            // B frag: two 16B LDS loads + shuffle.
            v8h b0 = *(const v8h*)&s_Bt[mn][kb0];
            v8h b1 = *(const v8h*)&s_Bt[mn][kb1];
            v16h bF = __builtin_shufflevector(b0, b1, 0, 1, 2, 3, 4, 5, 6, 7,
                                              8, 9, 10, 11, 12, 13, 14, 15);

            acc = __builtin_amdgcn_wmma_f32_16x16x32_f16(
                false, aF, false, bF, (short)0, acc, false, false);
            __syncthreads();
        }
    }

    // Epilogue: BN + SiLU, scatter per C/D layout.
    int pcol = w0 + mn;
    int mofs = (lane < 16) ? 0 : 8;
#pragma unroll
    for (int r = 0; r < 8; ++r) {
        int o = wave * 16 + r + mofs;
        float sc  = bn_g[o] * rsqrtf(bn_v[o] + EPS_);
        float val = (acc[r] - bn_m[o]) * sc + bn_b[o];
        val = val * sigmoidf_(val);
        out[((size_t)(b * CH_ + o) * H_ + h) * W_ + pcol] = val;
    }
}

// ---------------------------------------------------------------------------
// Fuse 1x1 conv (256->256) WMMA GEMM + BN + SiLU. B-chunks are staged into
// LDS with CDNA5 async global->LDS copies; each wave owns 2 o-tiles.
// ---------------------------------------------------------------------------
__global__ __launch_bounds__(256) void fuse_wmma_kernel(
    const _Float16* __restrict__ combh,  // (B,HW,256) f16, pixel-major
    const _Float16* __restrict__ wpf,    // packed f16 fuse weights
    const float* __restrict__ bn_g, const float* __restrict__ bn_b,
    const float* __restrict__ bn_m, const float* __restrict__ bn_v,
    float* __restrict__ out)             // (B,256,H,W)
{
    int tile = blockIdx.x;
    int wseg = tile & 3;
    int h    = (tile >> 2) & (H_ - 1);
    int b    = tile >> 8;
    int w0   = wseg * 16;

    int tid  = threadIdx.x;
    int lane = tid & 31;
    int wave = tid >> 5;

    const int mn  = lane & 15;
    const int kb0 = (lane < 16) ? 0 : 8;
    const int kb1 = (lane < 16) ? 16 : 24;
    const int pix0 = h * W_ + w0;

    __shared__ __align__(64) _Float16 s_Bt[16][32];   // [pixel][channel]
    const unsigned int ldsbase = lds_addr_of(&s_Bt[0][0]);

    v8f acc0 = {0.f, 0.f, 0.f, 0.f, 0.f, 0.f, 0.f, 0.f};
    v8f acc1 = acc0;

    for (int cc = 0; cc < C2_ / 32; ++cc) {
        // Async-stage 16 pixels x 32 channels (1KB) into LDS: waves 0-1,
        // 16 bytes per lane, tracked by ASYNCcnt.
        if (tid < 64) {
            int p = tid >> 2;   // pixel 0..15
            int q = tid & 3;    // 16B quarter of the 64B row
            const _Float16* g =
                combh + ((size_t)(b * HW_ + pix0 + p)) * C2_ + cc * 32 + q * 8;
            async_ld_b128(ldsbase + (unsigned int)(p * 64 + q * 16), g);
        }
        wait_async0();
        __syncthreads();

        v8h b0 = *(const v8h*)&s_Bt[mn][kb0];
        v8h b1 = *(const v8h*)&s_Bt[mn][kb1];
        v16h bF = __builtin_shufflevector(b0, b1, 0, 1, 2, 3, 4, 5, 6, 7,
                                          8, 9, 10, 11, 12, 13, 14, 15);

        v16h a0 = *(const v16h*)(wpf + (((size_t)(cc * 16 + 2 * wave) * 32 + lane) << 4));
        v16h a1 = *(const v16h*)(wpf + (((size_t)(cc * 16 + 2 * wave + 1) * 32 + lane) << 4));

        acc0 = __builtin_amdgcn_wmma_f32_16x16x32_f16(
            false, a0, false, bF, (short)0, acc0, false, false);
        acc1 = __builtin_amdgcn_wmma_f32_16x16x32_f16(
            false, a1, false, bF, (short)0, acc1, false, false);
        __syncthreads();
    }

    int pcol = w0 + mn;
    int mofs = (lane < 16) ? 0 : 8;
#pragma unroll
    for (int r = 0; r < 8; ++r) {
        int o0 = wave * 32 + r + mofs;
        int o1 = o0 + 16;
        float sc0 = bn_g[o0] * rsqrtf(bn_v[o0] + EPS_);
        float v0  = (acc0[r] - bn_m[o0]) * sc0 + bn_b[o0];
        v0 = v0 * sigmoidf_(v0);
        out[((size_t)(b * C2_ + o0) * H_ + h) * W_ + pcol] = v0;
        float sc1 = bn_g[o1] * rsqrtf(bn_v[o1] + EPS_);
        float v1  = (acc1[r] - bn_m[o1]) * sc1 + bn_b[o1];
        v1 = v1 * sigmoidf_(v1);
        out[((size_t)(b * C2_ + o1) * H_ + h) * W_ + pcol] = v1;
    }
}

// ---------------------------------------------------------------------------
// SCC attention helpers
// ---------------------------------------------------------------------------
__global__ void gap_kernel(const float* __restrict__ f, float* __restrict__ gap) {
    int bc = blockIdx.x;   // one block per (b, ch)
    const float* p = f + (size_t)bc * HW_;
    float s = 0.f;
    for (int i = threadIdx.x; i < HW_; i += 256) s += p[i];
    __shared__ float red[256];
    red[threadIdx.x] = s;
    __syncthreads();
    for (int st = 128; st > 0; st >>= 1) {
        if (threadIdx.x < st) red[threadIdx.x] += red[threadIdx.x + st];
        __syncthreads();
    }
    if (threadIdx.x == 0) gap[bc] = red[0] * (1.0f / (float)HW_);
}

__global__ void ca_kernel(const float* __restrict__ gap,
                          const float* __restrict__ w1,  // (R,CH)
                          const float* __restrict__ w2,  // (CH,R)
                          float* __restrict__ ca, int CH, int R) {
    int b = blockIdx.x;
    const float* g = gap + b * CH;
    __shared__ float hid[32];
    int t = threadIdx.x;
    if (t < R) {
        float s = 0.f;
        for (int c = 0; c < CH; ++c) s += g[c] * w1[t * CH + c];
        hid[t] = fmaxf(s, 0.f);
    }
    __syncthreads();
    if (t < CH) {
        float s = 0.f;
        for (int r = 0; r < R; ++r) s += hid[r] * w2[t * R + r];
        ca[b * CH + t] = sigmoidf_(s);
    }
}

__global__ void sp_kernel(const float* __restrict__ f,
                          const float* __restrict__ ca,
                          float* __restrict__ sp, int CH) {
    int idx = blockIdx.x * 256 + threadIdx.x;   // B*HW
    if (idx >= BN_ * HW_) return;
    int b = idx / HW_, p = idx % HW_;
    float sm = 0.f, mx = -3.4e38f;
    for (int c = 0; c < CH; ++c) {
        float v = f[((size_t)(b * CH + c)) * HW_ + p] * ca[b * CH + c];
        sm += v;
        mx = fmaxf(mx, v);
    }
    sp[(size_t)(b * 2 + 0) * HW_ + p] = sm / (float)CH;
    sp[(size_t)(b * 2 + 1) * HW_ + p] = mx;
}

__global__ void sa_kernel(const float* __restrict__ sp,
                          const float* __restrict__ w,   // (1,2,7,7)
                          float* __restrict__ sa) {
    int idx = blockIdx.x * 256 + threadIdx.x;   // B*HW
    if (idx >= BN_ * HW_) return;
    int b = idx / HW_, p = idx % HW_;
    int h = p / W_, x = p % W_;
    float s = 0.f;
    for (int c = 0; c < 2; ++c) {
        const float* sc = sp + (size_t)(b * 2 + c) * HW_;
        const float* wc = w + c * 49;
        for (int ky = 0; ky < 7; ++ky) {
            int yy = h + ky - 3;
            if (yy < 0 || yy >= H_) continue;
            for (int kx = 0; kx < 7; ++kx) {
                int xx = x + kx - 3;
                if (xx < 0 || xx >= W_) continue;
                s += sc[yy * W_ + xx] * wc[ky * 7 + kx];
            }
        }
    }
    sa[idx] = sigmoidf_(s);
}

__global__ void apply_kernel(const float* __restrict__ fin,
                             const float* __restrict__ ca,
                             const float* __restrict__ sa,
                             float* __restrict__ fout, int CH) {
    size_t idx = (size_t)blockIdx.x * 256 + threadIdx.x;  // B*CH*HW
    if (idx >= (size_t)BN_ * CH * HW_) return;
    int p = (int)(idx % HW_);
    size_t t = idx / HW_;
    int c = (int)(t % CH);
    int b = (int)(t / CH);
    fout[idx] = fin[idx] * ca[b * CH + c] * sa[b * HW_ + p];
}

// ---------------------------------------------------------------------------
// Launch
// ---------------------------------------------------------------------------
extern "C" void kernel_launch(void* const* d_in, const int* in_sizes, int n_in,
                              void* d_out, int out_size, void* d_ws, size_t ws_size,
                              hipStream_t stream) {
    const float* x      = (const float*)d_in[0];
    const float* off1_w = (const float*)d_in[1];
    const float* dcn1_w = (const float*)d_in[2];
    const float* bn1_g  = (const float*)d_in[3];
    const float* bn1_b  = (const float*)d_in[4];
    const float* bn1_m  = (const float*)d_in[5];
    const float* bn1_v  = (const float*)d_in[6];
    const float* ca1_w1 = (const float*)d_in[7];
    const float* ca1_w2 = (const float*)d_in[8];
    const float* sa1_w  = (const float*)d_in[9];
    const float* off2_w = (const float*)d_in[10];
    const float* dcn2_w = (const float*)d_in[11];
    const float* bn2_g  = (const float*)d_in[12];
    const float* bn2_b  = (const float*)d_in[13];
    const float* bn2_m  = (const float*)d_in[14];
    const float* bn2_v  = (const float*)d_in[15];
    const float* ca2_w1 = (const float*)d_in[16];
    const float* ca2_w2 = (const float*)d_in[17];
    const float* sa2_w  = (const float*)d_in[18];
    const float* fuse_w = (const float*)d_in[19];
    const float* bnf_g  = (const float*)d_in[20];
    const float* bnf_b  = (const float*)d_in[21];
    const float* bnf_m  = (const float*)d_in[22];
    const float* bnf_v  = (const float*)d_in[23];
    const float* caf_w1 = (const float*)d_in[24];
    const float* caf_w2 = (const float*)d_in[25];
    const float* saf_w  = (const float*)d_in[26];

    float* ws = (float*)d_ws;
    float* off1  = ws;                    // 294912
    float* off2  = off1 + 294912;         // 819200
    float* f1    = off2 + 819200;         // 2097152
    float* f2    = f1   + 2097152;        // 2097152
    float* fused = f2   + 2097152;        // 4194304
    float* gap   = fused + 4194304;       // 1024
    float* ca    = gap  + 1024;           // 1024
    float* sp    = ca   + 1024;           // 32768
    float* sa    = sp   + 32768;          // 16384
    _Float16* wp1   = (_Float16*)(sa + 16384);        // 147456 halves (73728 f)
    _Float16* wp2   = (_Float16*)(sa + 16384 + 73728);        // 409600 halves
    _Float16* wpf   = (_Float16*)(sa + 16384 + 73728 + 204800);   // 65536 halves
    _Float16* combh = (_Float16*)(sa + 16384 + 73728 + 204800 + 32768); // 4194304 halves

    const int npix   = BN_ * HW_;                 // 16384
    const int ptiles = BN_ * H_ * (W_ / 16);      // 1024

    // Weight repacks (cheap, deterministic, graph-capture safe)
    pack_dcn_w_kernel<3><<<(9 * 4 * 8 * 32 * 16 + 255) / 256, 256, 0, stream>>>(dcn1_w, wp1);
    pack_dcn_w_kernel<5><<<(25 * 4 * 8 * 32 * 16 + 255) / 256, 256, 0, stream>>>(dcn2_w, wp2);
    pack_fuse_w_kernel<<<(8 * 16 * 32 * 16 + 255) / 256, 256, 0, stream>>>(fuse_w, wpf);

    // Branch 1: offsets -> DCN(WMMA)+BN+SiLU -> SCC (in-place on f1)
    offset_conv_kernel<3><<<(BN_ * 18 * HW_ + 255) / 256, 256, 0, stream>>>(x, off1_w, off1, 18);
    dcn_wmma_kernel<3><<<ptiles, 256, 0, stream>>>(x, off1, wp1, bn1_g, bn1_b, bn1_m, bn1_v, f1);
    gap_kernel<<<BN_ * CH_, 256, 0, stream>>>(f1, gap);
    ca_kernel<<<BN_, CH_, 0, stream>>>(gap, ca1_w1, ca1_w2, ca, CH_, 8);
    sp_kernel<<<(npix + 255) / 256, 256, 0, stream>>>(f1, ca, sp, CH_);
    sa_kernel<<<(npix + 255) / 256, 256, 0, stream>>>(sp, sa1_w, sa);
    apply_kernel<<<(BN_ * CH_ * HW_ + 255) / 256, 256, 0, stream>>>(f1, ca, sa, f1, CH_);

    // Branch 2
    offset_conv_kernel<5><<<(BN_ * 50 * HW_ + 255) / 256, 256, 0, stream>>>(x, off2_w, off2, 50);
    dcn_wmma_kernel<5><<<ptiles, 256, 0, stream>>>(x, off2, wp2, bn2_g, bn2_b, bn2_m, bn2_v, f2);
    gap_kernel<<<BN_ * CH_, 256, 0, stream>>>(f2, gap);
    ca_kernel<<<BN_, CH_, 0, stream>>>(gap, ca2_w1, ca2_w2, ca, CH_, 8);
    sp_kernel<<<(npix + 255) / 256, 256, 0, stream>>>(f2, ca, sp, CH_);
    sa_kernel<<<(npix + 255) / 256, 256, 0, stream>>>(sp, sa2_w, sa);
    apply_kernel<<<(BN_ * CH_ * HW_ + 255) / 256, 256, 0, stream>>>(f2, ca, sa, f2, CH_);

    // Pack comb (f16, pixel-major), then fuse 1x1 conv (WMMA + async LDS)
    pack_comb_kernel<<<(BN_ * HW_ * C2_ + 255) / 256, 256, 0, stream>>>(f1, f2, combh);
    fuse_wmma_kernel<<<ptiles, 256, 0, stream>>>(combh, wpf, bnf_g, bnf_b, bnf_m, bnf_v, fused);

    // Final SCC -> d_out
    gap_kernel<<<BN_ * C2_, 256, 0, stream>>>(fused, gap);
    ca_kernel<<<BN_, C2_, 0, stream>>>(gap, caf_w1, caf_w2, ca, C2_, 16);
    sp_kernel<<<(npix + 255) / 256, 256, 0, stream>>>(fused, ca, sp, C2_);
    sa_kernel<<<(npix + 255) / 256, 256, 0, stream>>>(sp, saf_w, sa);
    apply_kernel<<<(BN_ * C2_ * HW_ + 255) / 256, 256, 0, stream>>>(fused, ca, sa, (float*)d_out, C2_);
}
